// CompactCSNN_17514876633112
// MI455X (gfx1250) — matrix operand order repaired and checked by reading, not compiled, not run on hardware
//
#include <hip/hip_runtime.h>

typedef __attribute__((ext_vector_type(16))) _Float16 v16h;
typedef __attribute__((ext_vector_type(8)))  _Float16 v8h;
typedef __attribute__((ext_vector_type(4)))  _Float16 v4h;
typedef __attribute__((ext_vector_type(8)))  float    v8f;
typedef __attribute__((ext_vector_type(4)))  float    f32x4;
typedef __attribute__((ext_vector_type(4)))  int      v4i;

#define BETA   0.9f
#define THRESH 1.0f
#define EPS    1e-5f

#define NB   128   // batch
#define HH   128   // input H=W
#define HO1  64    // conv1 out spatial
#define HO2  32    // conv2 out spatial
#define C1   16
#define C2   32
#define NCLS 10

#if __has_builtin(__builtin_amdgcn_global_load_async_to_lds_b128)
#define ASYNC_LDS 1
typedef __attribute__((address_space(1))) v4i* gptr_v4i;
typedef __attribute__((address_space(3))) v4i* lptr_v4i;
#endif

// ---------------------------------------------------------------- zero fill
__global__ void k_zero(float* __restrict__ p, size_t n) {
  size_t i = (size_t)blockIdx.x * blockDim.x + threadIdx.x;
  size_t stride = (size_t)gridDim.x * blockDim.x;
  for (; i < n; i += stride) p[i] = 0.f;
}

// -------------------------------------------- pack conv2 weights -> B fragments
// Layout: wfrag[((ntile*13 + chunk)*32 + lane)*16 + e], K = 32*chunk + (e<8 ? 8*(lane>>4)+e : 16+8*(lane>>4)+e-8)
// K = tap*16 + ic ; tap in [0,25) -> (ky,kx); K >= 400 zero-padded.
__global__ __launch_bounds__(256) void k_packw2(const float* __restrict__ w2,
                                                _Float16* __restrict__ wfrag) {
  int gid = blockIdx.x * 256 + threadIdx.x;
  if (gid >= 2 * 13 * 32) return;
  int ntile = gid / (13 * 32);
  int rem   = gid % (13 * 32);
  int lane  = rem % 32;
  int chunk = rem / 32;
  int n  = ntile * 16 + (lane & 15);
  int hi = lane >> 4;
  _Float16* dst = wfrag + (size_t)gid * 16;
#pragma unroll
  for (int e = 0; e < 16; ++e) {
    int K = 32 * chunk + (e < 8 ? 8 * hi + e : 16 + 8 * hi + (e - 8));
    float v = 0.f;
    if (K < 400) {
      int ic = K & 15, tap = K >> 4;
      int ky = tap / 5, kx = tap % 5;
      v = w2[((n * 16 + ic) * 5 + ky) * 5 + kx];   // OIHW
    }
    dst[e] = (_Float16)v;
  }
}

// ---------------- fused spike-gen + conv1 (direct, 1 in-ch) + BN1 stats
// cur1: NHWC [B][64][64][16] f32. Thread = one output pixel x 4 channels.
__global__ __launch_bounds__(256) void k_conv1(const float* __restrict__ x,
    const float* __restrict__ base, const float* __restrict__ rstep,
    const float* __restrict__ w1, float* __restrict__ cur1,
    float* __restrict__ bn1_sum, float* __restrict__ bn1_sqs) {
  __shared__ float sp[5][132];
  __shared__ float wsm[400];
  __shared__ float ssum[16], ssqs[16];
  const int tid = threadIdx.x;
  const int b  = blockIdx.x >> 6;
  const int oh = blockIdx.x & 63;
  if (tid < 400) wsm[tid] = w1[tid];
  if (tid < 16) { ssum[tid] = 0.f; ssqs[tid] = 0.f; }
  for (int i = tid; i < 5 * 132; i += 256) {
    int r = i / 132, j = i % 132;
    int ih = 2 * oh - 2 + r, iw = j - 2;
    float s = 0.f;
    if (ih >= 0 && ih < HH && iw >= 0 && iw < HH) {
      int idx = (b * HH + ih) * HH + iw;
      float xn = fminf(fmaxf((x[idx] + 1.f) * 0.5f, 0.f), 1.f);
      float sr = 0.5f * base[idx] + 0.5f * rstep[idx];
      s = (sr < xn) ? 1.f : 0.f;
    }
    sp[r][j] = s;
  }
  __syncthreads();
  const int ow = tid >> 2;          // 0..63
  const int ch = (tid & 3) * 4;     // 0,4,8,12
  f32x4 acc = {};
#pragma unroll
  for (int ky = 0; ky < 5; ++ky)
#pragma unroll
    for (int kx = 0; kx < 5; ++kx) {
      float s = sp[ky][2 * ow + kx];
      int wb = ky * 5 + kx;
#pragma unroll
      for (int c = 0; c < 4; ++c)
        acc[c] = fmaf(s, wsm[(ch + c) * 25 + wb], acc[c]);
    }
  *(f32x4*)(cur1 + (size_t)((b * HO1 + oh) * HO1 + ow) * 16 + ch) = acc;
#pragma unroll
  for (int c = 0; c < 4; ++c) {
    atomicAdd(&ssum[ch + c], acc[c]);
    atomicAdd(&ssqs[ch + c], acc[c] * acc[c]);
  }
  __syncthreads();
  if (tid < 16) {
    atomicAdd(&bn1_sum[tid], ssum[tid]);
    atomicAdd(&bn1_sqs[tid], ssqs[tid]);
  }
}

// ---------------- BN finalize: per-channel scale/shift from batch stats
__global__ void k_bnfin(const float* __restrict__ sum, const float* __restrict__ sqs,
                        const float* __restrict__ g, const float* __restrict__ bb,
                        float* __restrict__ scale, float* __restrict__ shift,
                        int n, float cinv) {
  int c = threadIdx.x;
  if (c < n) {
    float mean = sum[c] * cinv;
    float var  = sqs[c] * cinv - mean * mean;
    float sc   = g[c] * rsqrtf(var + EPS);
    scale[c] = sc;
    shift[c] = bb[c] - mean * sc;
  }
}

// ---------------- LIF1 (x4 vectorized): BN apply + leaky update; spikes -> f16 NHWC
__global__ __launch_bounds__(256) void k_lif1(const float* __restrict__ cur1,
    const float* __restrict__ scale, const float* __restrict__ shift,
    float* __restrict__ mem1, _Float16* __restrict__ spk1) {
  size_t g = ((size_t)blockIdx.x * 256 + threadIdx.x) * 4;
  int ch = (int)(g & 15);                     // 0,4,8,12 -> 4 consecutive channels
  f32x4 cr = *(const f32x4*)(cur1 + g);
  f32x4 sc = *(const f32x4*)(scale + ch);
  f32x4 sh = *(const f32x4*)(shift + ch);
  f32x4 m  = *(const f32x4*)(mem1 + g);
  f32x4 mn;
  v4h s;
#pragma unroll
  for (int j = 0; j < 4; ++j) {
    float cur = fmaf(cr[j], sc[j], sh[j]);
    float reset = (m[j] > THRESH) ? 1.f : 0.f;
    mn[j] = BETA * m[j] + cur - reset * THRESH;
    s[j] = (mn[j] > THRESH) ? (_Float16)1.0f : (_Float16)0.0f;
  }
  *(f32x4*)(mem1 + g) = mn;
  *(v4h*)(spk1 + g) = s;
}

// ---------------- conv2 as implicit GEMM on WMMA + BN2 stats
// spk1: NHWC f16 [B][64][64][16]; cur2: NCHW f32 [B][32][32][32]
// block = 128 threads (4 waves), one (b, oh) row per block.
// wave task: ow-tile (0/16) x ntile (out-ch 0-15 / 16-31); K = 400 over 13 chunks.
__global__ __launch_bounds__(128) void k_conv2(const _Float16* __restrict__ spk1,
    const _Float16* __restrict__ wfrag, float* __restrict__ cur2,
    float* __restrict__ bn2_sum, float* __restrict__ bn2_sqs) {
  __shared__ __align__(16) _Float16 tile[5 * 1088];  // [row 5][col 68][ic 16]
  __shared__ float ssum[32], ssqs[32];
  const int tid = threadIdx.x;
  const int b  = blockIdx.x >> 5;
  const int oh = blockIdx.x & 31;
  if (tid < 32) { ssum[tid] = 0.f; ssqs[tid] = 0.f; }
  // stage 5 input rows (64 cols x 16 ch) + zero pads into LDS
  for (int i = tid; i < 680; i += 128) {
    if (i < 640) {
      int r = i >> 7, q = i & 127;
      int ih = 2 * oh - 2 + r;
      int off = r * 1088 + 32 + q * 8;
      if (ih >= 0 && ih < HO1) {
        const _Float16* src = spk1 + ((size_t)(b * HO1 + ih) * HO1) * 16 + q * 8;
#ifdef ASYNC_LDS
        // memory -> LDS without VGPR round-trip (ASYNCcnt-tracked)
        __builtin_amdgcn_global_load_async_to_lds_b128(
            (gptr_v4i)src, (lptr_v4i)(tile + off), 0, 0);
#else
        *(uint4*)(tile + off) = *(const uint4*)src;
#endif
      } else {
        *(uint4*)(tile + off) = make_uint4(0u, 0u, 0u, 0u);
      }
    } else {
      int p = i - 640;                 // 40 pad vectors (left/right halo)
      int r = p / 8, q = p & 7;
      int off = r * 1088 + (q < 4 ? q * 8 : 1056 + (q - 4) * 8);
      *(uint4*)(tile + off) = make_uint4(0u, 0u, 0u, 0u);
    }
  }
#ifdef ASYNC_LDS
#if __has_builtin(__builtin_amdgcn_s_wait_asynccnt)
  __builtin_amdgcn_s_wait_asynccnt(0);
#else
  asm volatile("s_wait_asynccnt 0" ::: "memory");
#endif
#endif
  __syncthreads();

  const int wave  = tid >> 5;
  const int lane  = tid & 31;
  const int ow0   = (wave & 1) * 16;
  const int ntile = wave >> 1;
  const int m  = lane & 15;
  const int hi = lane >> 4;
  const int jb = 2 * ow0 + 2 * m;      // LDS col base for this output position

  v8f acc = {};
  const _Float16* wf = wfrag + (size_t)(ntile * 13) * 512 + (size_t)lane * 16;
#pragma unroll
  for (int c = 0; c < 13; ++c) {
    int t0 = 2 * c;
    int t1 = (t0 + 1 < 25) ? t0 + 1 : 24;     // tail tap: B weights are zero there
    int ky0 = t0 / 5, kx0 = t0 % 5;
    int ky1 = t1 / 5, kx1 = t1 % 5;
    v8h a0 = *(const v8h*)(tile + ky0 * 1088 + (jb + kx0) * 16 + hi * 8);
    v8h a1 = *(const v8h*)(tile + ky1 * 1088 + (jb + kx1) * 16 + hi * 8);
    v16h a;
#pragma unroll
    for (int k = 0; k < 8; ++k) { a[k] = a0[k]; a[k + 8] = a1[k]; }
    v16h bf = *(const v16h*)(wf + (size_t)c * 512);
    acc = __builtin_amdgcn_wmma_f32_16x16x32_f16(false, a, false, bf,
                                                 (short)0, acc, false, false);
  }
  // epilogue: lane owns channel ch, 8 consecutive ow positions
  const int ch = ntile * 16 + (lane & 15);
  const int owbase = ow0 + hi * 8;
  float* dst = cur2 + (((size_t)(b * C2 + ch) * HO2 + oh) * HO2 + owbase);
  float lsum = 0.f, lsqs = 0.f;
#pragma unroll
  for (int i = 0; i < 8; ++i) {
    float v = acc[i];
    lsum += v; lsqs += v * v;
    dst[i] = v;
  }
  atomicAdd(&ssum[ch], lsum);
  atomicAdd(&ssqs[ch], lsqs);
  __syncthreads();
  if (tid < 32) {
    atomicAdd(&bn2_sum[tid], ssum[tid]);
    atomicAdd(&bn2_sqs[tid], ssqs[tid]);
  }
}

// ---------------- LIF2 (x4 vectorized) + global-avg-pool
// NCHW: one block (256 thr x 4 elem) = exactly one (b,c) plane of 1024 px
__global__ __launch_bounds__(256) void k_lif2(const float* __restrict__ cur2,
    const float* __restrict__ scale, const float* __restrict__ shift,
    float* __restrict__ mem2, float* __restrict__ pool) {
  __shared__ float red[256];
  size_t g = ((size_t)blockIdx.x * 256 + threadIdx.x) * 4;
  int ch = (int)((g >> 10) & 31);             // uniform across block
  f32x4 cr = *(const f32x4*)(cur2 + g);
  f32x4 m  = *(const f32x4*)(mem2 + g);
  float sc = scale[ch], sh = shift[ch];
  f32x4 mn;
  float lsum = 0.f;
#pragma unroll
  for (int j = 0; j < 4; ++j) {
    float cur = fmaf(cr[j], sc, sh);
    float reset = (m[j] > THRESH) ? 1.f : 0.f;
    mn[j] = BETA * m[j] + cur - reset * THRESH;
    lsum += (mn[j] > THRESH) ? 1.f : 0.f;
  }
  *(f32x4*)(mem2 + g) = mn;
  red[threadIdx.x] = lsum;
  __syncthreads();
  for (int s = 128; s > 0; s >>= 1) {
    if (threadIdx.x < s) red[threadIdx.x] += red[threadIdx.x + s];
    __syncthreads();
  }
  if (threadIdx.x == 0) pool[blockIdx.x] = red[0];   // block == (b,c): no atomics
}

// ---------------- head: pool/1024 -> FC(32->10) -> LIF3 -> spikes out
__global__ __launch_bounds__(128) void k_head(const float* __restrict__ pool,
    const float* __restrict__ fcw, float* __restrict__ mem3,
    float* __restrict__ out_t) {
  __shared__ float fw[NCLS * C2];
  int tid = threadIdx.x;
  for (int i = tid; i < NCLS * C2; i += 128) fw[i] = fcw[i];
  __syncthreads();
  const float inv = 1.0f / 1024.0f;
  int b = tid;
#pragma unroll
  for (int c = 0; c < NCLS; ++c) {
    float cur = 0.f;
#pragma unroll
    for (int ic = 0; ic < C2; ++ic)
      cur = fmaf(pool[b * C2 + ic] * inv, fw[c * C2 + ic], cur);
    float m = mem3[b * NCLS + c];
    float reset = (m > THRESH) ? 1.f : 0.f;
    float mn = BETA * m + cur - reset * THRESH;
    mem3[b * NCLS + c] = mn;
    out_t[b * NCLS + c] = (mn > THRESH) ? 1.f : 0.f;
  }
}

extern "C" void kernel_launch(void* const* d_in, const int* in_sizes, int n_in,
                              void* d_out, int out_size, void* d_ws, size_t ws_size,
                              hipStream_t stream) {
  (void)in_sizes; (void)n_in; (void)out_size; (void)ws_size;
  const float* x     = (const float*)d_in[0];
  const float* base  = (const float*)d_in[1];
  const float* srand = (const float*)d_in[2];
  // d_in[3] = num_steps (fixed at 16 per reference)
  const float* w1    = (const float*)d_in[4];
  const float* g1    = (const float*)d_in[5];
  const float* b1    = (const float*)d_in[6];
  const float* w2    = (const float*)d_in[7];
  const float* g2    = (const float*)d_in[8];
  const float* b2    = (const float*)d_in[9];
  const float* fcw   = (const float*)d_in[10];
  float* out = (float*)d_out;

  float* ws = (float*)d_ws;
  float* mem1 = ws;                           // 8388608 f
  float* mem2 = mem1 + 8388608;               // 4194304 f
  float* mem3 = mem2 + 4194304;               // 1280 f
  float* pool = mem3 + 1280;                  // 4096 f (fully overwritten each step)
  float* bn1_sum = pool + 4096;               // 16   (per-step zero region: 96 f)
  float* bn1_sqs = bn1_sum + 16;              // 16
  float* bn2_sum = bn1_sqs + 16;              // 32
  float* bn2_sqs = bn2_sum + 32;              // 32
  float* bn1_scale = bn2_sqs + 32;            // 16
  float* bn1_shift = bn1_scale + 16;          // 16
  float* bn2_scale = bn1_shift + 16;          // 32
  float* bn2_shift = bn2_scale + 32;          // 32
  float* cur1 = bn2_shift + 32;               // 8388608 f
  _Float16* spk1 = (_Float16*)(cur1 + 8388608);   // 8388608 halfs
  float* cur2 = (float*)(spk1 + 8388608);     // 4194304 f
  _Float16* wfrag = (_Float16*)(cur2 + 4194304);  // 13312 halfs

  // zero LIF membrane states (mem1|mem2|mem3 contiguous), pack conv2 weights
  k_zero<<<4096, 256, 0, stream>>>(mem1, (size_t)8388608 + 4194304 + 1280);
  k_packw2<<<4, 256, 0, stream>>>(w2, wfrag);

  const size_t stepsz = (size_t)NB * HH * HH;
  for (int t = 0; t < 16; ++t) {
    k_zero<<<1, 96, 0, stream>>>(bn1_sum, 96);
    k_conv1<<<NB * HO1, 256, 0, stream>>>(x, base, srand + (size_t)t * stepsz,
                                          w1, cur1, bn1_sum, bn1_sqs);
    k_bnfin<<<1, 16, 0, stream>>>(bn1_sum, bn1_sqs, g1, b1, bn1_scale, bn1_shift,
                                  16, 1.0f / 524288.0f);
    k_lif1<<<8192, 256, 0, stream>>>(cur1, bn1_scale, bn1_shift, mem1, spk1);
    k_conv2<<<NB * HO2, 128, 0, stream>>>(spk1, wfrag, cur2, bn2_sum, bn2_sqs);
    k_bnfin<<<1, 32, 0, stream>>>(bn2_sum, bn2_sqs, g2, b2, bn2_scale, bn2_shift,
                                  32, 1.0f / 131072.0f);
    k_lif2<<<4096, 256, 0, stream>>>(cur2, bn2_scale, bn2_shift, mem2, pool);
    k_head<<<1, 128, 0, stream>>>(pool, fcw, mem3, out + (size_t)t * NB * NCLS);
  }
}
